// GNNAdModel_18236431139064
// MI455X (gfx1250) — matrix operand order repaired and checked by reading, not compiled, not run on hardware
//
#include <hip/hip_runtime.h>
#include <hip/hip_bf16.h>
#include <math.h>

// ---------------------------------------------------------------------------
// GAT link-prediction model for MI455X (gfx1250, wave32, WMMA).
// Dense GEMMs use v_wmma_f32_16x16x32_f16 (f16 in, f32 accumulate).
// Edge softmax/scatter uses native f32 global atomics.
// ---------------------------------------------------------------------------

typedef __attribute__((ext_vector_type(16))) _Float16 v16h;
typedef __attribute__((ext_vector_type(8)))  float    v8f;

#define NN   100000
#define EE   1600000
#define DD   64
#define HH   4
#define OO   32
#define QQ   100000

// ------------------------------ utility -----------------------------------

__global__ void fill_f32(float* __restrict__ p, int n) {
  int i = blockIdx.x * blockDim.x + threadIdx.x;
  if (i < n) p[i] = 0.0f;
}

// W [Hh, K, O] f32  ->  Bt [(Hh*O), K] f16  (Bt[h*O+o][k] = W[h][k][o])
__global__ void prep_weight(const float* __restrict__ W, _Float16* __restrict__ Bt,
                            int Hh, int K, int O) {
  int idx = blockIdx.x * blockDim.x + threadIdx.x;
  int total = Hh * K * O;
  if (idx >= total) return;
  int k = idx % K;
  int t = idx / K;
  int o = t % O;
  int h = t / O;
  Bt[(size_t)(h * O + o) * K + k] = (_Float16)W[((size_t)(h * K + k)) * O + o];
}

// ------------------------------ encoder ------------------------------------
// h = ReLU(LN(x @ W_enc[type] + b_enc[type]))  (only the selected type)
// also h_res = h @ W_res + b_res
__global__ void encoder_kernel(const float* __restrict__ x,     // [N,64]
                               const int*   __restrict__ types, // [N]
                               const float* __restrict__ W_enc, // [3,64,64] (i,o)
                               const float* __restrict__ b_enc, // [3,64]
                               const float* __restrict__ ln_g,  // [3,64]
                               const float* __restrict__ ln_b,  // [3,64]
                               const float* __restrict__ W_res, // [64,32]
                               const float* __restrict__ b_res, // [32]
                               _Float16* __restrict__ h16,      // [N,64] packed
                               float* __restrict__ hres) {      // [N,32]
  int nidx = blockIdx.x * blockDim.x + threadIdx.x;
  if (nidx >= NN) return;
  int t = types[nidx];
  const float* Wt = W_enc + (size_t)t * 64 * 64;

  float z[64];
  #pragma unroll
  for (int o = 0; o < 64; ++o) z[o] = b_enc[t * 64 + o];
  for (int i = 0; i < 64; ++i) {
    float xv = x[(size_t)nidx * 64 + i];
    const float* wr = Wt + i * 64;
    #pragma unroll
    for (int o = 0; o < 64; ++o) z[o] = fmaf(xv, wr[o], z[o]);
  }
  // LayerNorm
  float s = 0.f, sq = 0.f;
  #pragma unroll
  for (int o = 0; o < 64; ++o) { s += z[o]; sq += z[o] * z[o]; }
  float mu = s * (1.0f / 64.0f);
  float var = fmaxf(sq * (1.0f / 64.0f) - mu * mu, 0.0f);
  float inv = rsqrtf(var + 1e-5f);
  float h[64];
  #pragma unroll
  for (int o = 0; o < 64; ++o) {
    float v = (z[o] - mu) * inv * ln_g[t * 64 + o] + ln_b[t * 64 + o];
    v = fmaxf(v, 0.0f);
    h[o] = v;
    h16[(size_t)nidx * 64 + o] = (_Float16)v;
  }
  // residual projection
  float r[32];
  #pragma unroll
  for (int o = 0; o < 32; ++o) r[o] = b_res[o];
  for (int i = 0; i < 64; ++i) {
    float hv = h[i];
    const float* wr = W_res + i * 32;
    #pragma unroll
    for (int o = 0; o < 32; ++o) r[o] = fmaf(hv, wr[o], r[o]);
  }
  #pragma unroll
  for (int o = 0; o < 32; ++o) hres[(size_t)nidx * 32 + o] = r[o];
}

// ------------------------------ WMMA GEMM ----------------------------------
// C[M,Nc] = act(A[M,K] @ Bt[Nc,K]^T + bias)
// A row-major f16, Bt row-major f16 (i.e. B stored K-contiguous per column).
// act: 0 = none, 1 = relu. Writes f32 (C) and/or f16 (Ch).
// M % 16 == 0, K % 32 == 0, Nc % 16 == 0.
__global__ void wmma_gemm_kernel(const _Float16* __restrict__ A,
                                 const _Float16* __restrict__ Bt,
                                 const float* __restrict__ bias,
                                 float* __restrict__ C,
                                 _Float16* __restrict__ Ch,
                                 int M, int K, int Nc, int act) {
  int wave = threadIdx.x >> 5;            // 4 waves / block
  int lane = threadIdx.x & 31;
  int tileM = blockIdx.x * 4 + wave;
  int tileN = blockIdx.y;
  if (tileM * 16 >= M) return;

  int m   = lane & 15;
  int row = tileM * 16 + m;               // A source row for this lane
  int n   = tileN * 16 + (lane & 15);     // B column / C column for this lane
  int kba = (lane < 16) ? 0 : 8;          // A K-sub-base (16-bit A 16x32 layout)
  int kbb = (lane < 16) ? 0 : 8;          // B K-sub-base (same K order as A)

  v8f acc = {};
  for (int k0 = 0; k0 < K; k0 += 32) {
    v16h a, b;
    const _Float16* Ar = A + (size_t)row * K + k0;
    #pragma unroll
    for (int v = 0; v < 8; ++v) {
      int kk = ((v >> 2) * 16) + kba + 2 * (v & 3);
      a[2 * v]     = Ar[kk];
      a[2 * v + 1] = Ar[kk + 1];
    }
    const _Float16* Br = Bt + (size_t)n * K + k0;
    #pragma unroll
    for (int v = 0; v < 8; ++v) {
      int kk = ((v >> 2) * 16) + kbb + 2 * (v & 3);
      b[2 * v]     = Br[kk];
      b[2 * v + 1] = Br[kk + 1];
    }
    acc = __builtin_amdgcn_wmma_f32_16x16x32_f16(
        false, a, false, b, (short)0, acc, false, false);
  }

  int mBase = (lane < 16) ? 0 : 8;
  float bv = bias ? bias[n] : 0.0f;
  #pragma unroll
  for (int r = 0; r < 8; ++r) {
    int rr = tileM * 16 + mBase + r;
    float v = acc[r] + bv;
    if (act == 1) v = fmaxf(v, 0.0f);
    if (C)  C[(size_t)rr * Nc + n]  = v;
    if (Ch) Ch[(size_t)rr * Nc + n] = (_Float16)v;
  }
}

// ---------------------------- attention scores -----------------------------
__global__ void attn_score_kernel(const float* __restrict__ hp,   // [N,128]
                                  const float* __restrict__ a,    // [4,64]
                                  float* __restrict__ ssrc,       // [N,4]
                                  float* __restrict__ sdst) {     // [N,4]
  int idx = blockIdx.x * blockDim.x + threadIdx.x;
  if (idx >= NN * HH) return;
  int node = idx >> 2, h = idx & 3;
  const float* hpr = hp + (size_t)node * 128 + h * 32;
  const float* as  = a + h * 64;
  float s1 = 0.f, s2 = 0.f;
  #pragma unroll
  for (int o = 0; o < 32; ++o) {
    s1 = fmaf(hpr[o], as[o], s1);
    s2 = fmaf(hpr[o], as[32 + o], s2);
  }
  ssrc[idx] = s1;
  sdst[idx] = s2;
}

// ------------------------------ edge passes --------------------------------
__device__ __forceinline__ float edge_score(const float* ssrc, const float* sdst,
                                            const float* ew, int s, int d,
                                            int edge, int h) {
  float v = ssrc[s * 4 + h] + sdst[d * 4 + h];
  v = (v > 0.0f) ? v : 0.2f * v;                 // leaky_relu 0.2
  return v + log1pf(fmaxf(ew[edge], 0.0f));      // + ew_log
}

// pass 1: segment max into emax (emax pre-filled 0; max-with-0 baseline)
__global__ void edge_max_kernel(const int* __restrict__ eidx,
                                const float* __restrict__ ew,
                                const float* __restrict__ ssrc,
                                const float* __restrict__ sdst,
                                float* __restrict__ emax) {
  int idx = blockIdx.x * blockDim.x + threadIdx.x;
  if (idx >= EE * HH) return;
  int edge = idx >> 2, h = idx & 3;
  int s = eidx[edge], d = eidx[EE + edge];
  float v = edge_score(ssrc, sdst, ew, s, d, edge, h);
  if (v > 0.0f)   // nonneg floats compare identically in u32 bit order
    atomicMax((unsigned int*)&emax[d * 4 + h], __float_as_uint(v));
}

// pass 2: exp(e - max), store, and segment sum
__global__ void edge_exp_kernel(const int* __restrict__ eidx,
                                const float* __restrict__ ew,
                                const float* __restrict__ ssrc,
                                const float* __restrict__ sdst,
                                const float* __restrict__ emax,
                                float* __restrict__ expe,
                                float* __restrict__ sum) {
  int idx = blockIdx.x * blockDim.x + threadIdx.x;
  if (idx >= EE * HH) return;
  int edge = idx >> 2, h = idx & 3;
  int s = eidx[edge], d = eidx[EE + edge];
  float v = edge_score(ssrc, sdst, ew, s, d, edge, h);
  float ex = expf(v - emax[d * 4 + h]);
  expe[idx] = ex;
  atomicAdd(&sum[d * 4 + h], ex);
}

// pass 3: h_new[dst] += alpha * hp[src]
__global__ void edge_scatter_kernel(const int* __restrict__ eidx,
                                    const float* __restrict__ expe,
                                    const float* __restrict__ sum,
                                    const float* __restrict__ hp,
                                    float* __restrict__ hnew) {
  int idx = blockIdx.x * blockDim.x + threadIdx.x;
  if (idx >= EE * HH) return;
  int edge = idx >> 2, h = idx & 3;
  int s = eidx[edge], d = eidx[EE + edge];
  float alpha = expe[idx] / (sum[d * 4 + h] + 1e-8f);
  const float* hps = hp + (size_t)s * 128 + h * 32;
  float* outp = hnew + (size_t)d * 128 + h * 32;
  #pragma unroll
  for (int o = 0; o < 32; ++o) atomicAdd(&outp[o], alpha * hps[o]);
}

// ELU + convert to f16 (layers 0,1 feed the next GEMM)
__global__ void finalize01_kernel(const float* __restrict__ hnew,
                                  _Float16* __restrict__ h16) {
  int idx = blockIdx.x * blockDim.x + threadIdx.x;
  if (idx >= NN * 128) return;
  float v = hnew[idx];
  v = (v > 0.0f) ? v : expm1f(v);
  h16[idx] = (_Float16)v;
}

// layer 2: mean over heads of elu(h_new) + h_res  ->  h3 [N,32]
__global__ void finalize2_kernel(const float* __restrict__ hnew,
                                 const float* __restrict__ hres,
                                 float* __restrict__ h3) {
  int idx = blockIdx.x * blockDim.x + threadIdx.x;
  if (idx >= NN * 32) return;
  int node = idx >> 5, o = idx & 31;
  float acc = 0.f;
  #pragma unroll
  for (int h = 0; h < HH; ++h) {
    float v = hnew[(size_t)node * 128 + h * 32 + o];
    acc += (v > 0.0f) ? v : expm1f(v);
  }
  h3[idx] = acc * (1.0f / HH) + hres[idx];
}

// pair features: [h3[q0] || h3[q1]] -> f16 [Q,64]
__global__ void pair_gather_kernel(const int* __restrict__ qp,
                                   const float* __restrict__ h3,
                                   _Float16* __restrict__ pair16) {
  int idx = blockIdx.x * blockDim.x + threadIdx.x;
  if (idx >= QQ * 64) return;
  int q = idx >> 6, j = idx & 63;
  int node = (j < 32) ? qp[q * 2] : qp[q * 2 + 1];
  int o = j & 31;
  pair16[idx] = (_Float16)h3[(size_t)node * 32 + o];
}

// final 64->1 + sigmoid
__global__ void head_kernel(const float* __restrict__ x2,  // [Q,64]
                            const float* __restrict__ w,   // [64]
                            const float* __restrict__ b,   // [1]
                            float* __restrict__ out) {
  int q = blockIdx.x * blockDim.x + threadIdx.x;
  if (q >= QQ) return;
  const float* xr = x2 + (size_t)q * 64;
  float s = b[0];
  #pragma unroll
  for (int i = 0; i < 64; ++i) s = fmaf(xr[i], w[i], s);
  out[q] = 1.0f / (1.0f + expf(-s));
}

// ------------------------------- launcher ----------------------------------

static inline int cdiv(int a, int b) { return (a + b - 1) / b; }

extern "C" void kernel_launch(void* const* d_in, const int* in_sizes, int n_in,
                              void* d_out, int out_size, void* d_ws, size_t ws_size,
                              hipStream_t stream) {
  (void)in_sizes; (void)n_in; (void)out_size; (void)ws_size;

  const float* x      = (const float*)d_in[0];
  const int*   types  = (const int*)  d_in[1];
  const int*   eidx   = (const int*)  d_in[2];   // [2,E]
  const float* ew     = (const float*)d_in[3];
  const int*   qp     = (const int*)  d_in[4];   // [Q,2]
  const float* W_enc  = (const float*)d_in[5];
  const float* b_enc  = (const float*)d_in[6];
  const float* ln_g   = (const float*)d_in[7];
  const float* ln_b   = (const float*)d_in[8];
  const float* W_g0   = (const float*)d_in[9];
  const float* a_g0   = (const float*)d_in[10];
  const float* W_g1   = (const float*)d_in[11];
  const float* a_g1   = (const float*)d_in[12];
  const float* W_g2   = (const float*)d_in[13];
  const float* a_g2   = (const float*)d_in[14];
  const float* W_res  = (const float*)d_in[15];
  const float* b_res  = (const float*)d_in[16];
  const float* W_p1   = (const float*)d_in[17];
  const float* b_p1   = (const float*)d_in[18];
  const float* W_p2   = (const float*)d_in[19];
  const float* b_p2   = (const float*)d_in[20];
  const float* W_p3   = (const float*)d_in[21];
  const float* b_p3   = (const float*)d_in[22];
  float* out = (float*)d_out;

  char* ws = (char*)d_ws;
  size_t off = 0;
  auto alloc = [&](size_t bytes) { size_t o = off; off += (bytes + 255) & ~(size_t)255; return o; };
  size_t o_hres   = alloc((size_t)NN * 32 * 4);
  size_t o_hcur16 = alloc((size_t)NN * 128 * 2);   // layer input f16; later aliased as pair16
  size_t o_hp     = alloc((size_t)NN * 128 * 4);
  size_t o_ssrc   = alloc((size_t)NN * 4 * 4);
  size_t o_sdst   = alloc((size_t)NN * 4 * 4);
  size_t o_emax   = alloc((size_t)NN * 4 * 4);
  size_t o_sum    = alloc((size_t)NN * 4 * 4);
  size_t o_expe   = alloc((size_t)EE * 4 * 4);     // later aliased as x1_16
  size_t o_hnew   = alloc((size_t)NN * 128 * 4);   // later aliased as x2_32
  size_t o_h3     = alloc((size_t)NN * 32 * 4);
  size_t o_bt0    = alloc(128 * 64 * 2);
  size_t o_bt1    = alloc(128 * 128 * 2);
  size_t o_bt2    = alloc(128 * 128 * 2);
  size_t o_btp1   = alloc(128 * 64 * 2);
  size_t o_btp2   = alloc(64 * 128 * 2);

  float*     hres   = (float*)(ws + o_hres);
  _Float16*  hcur16 = (_Float16*)(ws + o_hcur16);
  float*     hp     = (float*)(ws + o_hp);
  float*     ssrc   = (float*)(ws + o_ssrc);
  float*     sdst   = (float*)(ws + o_sdst);
  float*     emax   = (float*)(ws + o_emax);
  float*     sum    = (float*)(ws + o_sum);
  float*     expe   = (float*)(ws + o_expe);
  float*     hnew   = (float*)(ws + o_hnew);
  float*     h3     = (float*)(ws + o_h3);
  _Float16*  bt0    = (_Float16*)(ws + o_bt0);
  _Float16*  bt1    = (_Float16*)(ws + o_bt1);
  _Float16*  bt2    = (_Float16*)(ws + o_bt2);
  _Float16*  btp1   = (_Float16*)(ws + o_btp1);
  _Float16*  btp2   = (_Float16*)(ws + o_btp2);
  _Float16*  pair16 = hcur16;            // alias (free after layer-2 GEMM)
  _Float16*  x1_16  = (_Float16*)expe;   // alias (free after layer-2 edges)
  float*     x2_32  = hnew;              // alias (free after finalize2)

  const int TB = 256;

  // weight prep (f32 -> transposed f16)
  prep_weight<<<cdiv(4 * 64 * 32, TB), TB, 0, stream>>>(W_g0, bt0, 4, 64, 32);
  prep_weight<<<cdiv(4 * 128 * 32, TB), TB, 0, stream>>>(W_g1, bt1, 4, 128, 32);
  prep_weight<<<cdiv(4 * 128 * 32, TB), TB, 0, stream>>>(W_g2, bt2, 4, 128, 32);
  prep_weight<<<cdiv(64 * 128, TB), TB, 0, stream>>>(W_p1, btp1, 1, 64, 128);
  prep_weight<<<cdiv(128 * 64, TB), TB, 0, stream>>>(W_p2, btp2, 1, 128, 64);

  // encoder (+ fused residual projection)
  encoder_kernel<<<cdiv(NN, 128), 128, 0, stream>>>(
      x, types, W_enc, b_enc, ln_g, ln_b, W_res, b_res, hcur16, hres);

  const _Float16* bts[3] = {bt0, bt1, bt2};
  const float*    ags[3] = {a_g0, a_g1, a_g2};
  const int       Ks[3]  = {64, 128, 128};

  dim3 gemmBlk(128);
  for (int l = 0; l < 3; ++l) {
    // hp = h @ Wcat   (N x K @ K x 128), WMMA
    dim3 gemmGrd(cdiv(NN / 16, 4), 128 / 16);
    wmma_gemm_kernel<<<gemmGrd, gemmBlk, 0, stream>>>(
        hcur16, bts[l], nullptr, hp, nullptr, NN, Ks[l], 128, 0);

    attn_score_kernel<<<cdiv(NN * HH, TB), TB, 0, stream>>>(hp, ags[l], ssrc, sdst);

    fill_f32<<<cdiv(NN * 4, TB), TB, 0, stream>>>(emax, NN * 4);
    fill_f32<<<cdiv(NN * 4, TB), TB, 0, stream>>>(sum, NN * 4);
    fill_f32<<<cdiv(NN * 128, TB), TB, 0, stream>>>(hnew, NN * 128);

    edge_max_kernel<<<cdiv(EE * HH, TB), TB, 0, stream>>>(eidx, ew, ssrc, sdst, emax);
    edge_exp_kernel<<<cdiv(EE * HH, TB), TB, 0, stream>>>(eidx, ew, ssrc, sdst, emax, expe, sum);
    edge_scatter_kernel<<<cdiv(EE * HH, TB), TB, 0, stream>>>(eidx, expe, sum, hp, hnew);

    if (l < 2) {
      finalize01_kernel<<<cdiv(NN * 128, TB), TB, 0, stream>>>(hnew, hcur16);
    } else {
      finalize2_kernel<<<cdiv(NN * 32, TB), TB, 0, stream>>>(hnew, hres, h3);
    }
  }

  // pair MLP head
  pair_gather_kernel<<<cdiv(QQ * 64, TB), TB, 0, stream>>>(qp, h3, pair16);
  {
    dim3 g1(cdiv(QQ / 16, 4), 128 / 16);
    wmma_gemm_kernel<<<g1, gemmBlk, 0, stream>>>(pair16, btp1, b_p1,
                                                 nullptr, x1_16, QQ, 64, 128, 1);
    dim3 g2(cdiv(QQ / 16, 4), 64 / 16);
    wmma_gemm_kernel<<<g2, gemmBlk, 0, stream>>>(x1_16, btp2, b_p2,
                                                 x2_32, nullptr, QQ, 128, 64, 1);
  }
  head_kernel<<<cdiv(QQ, TB), TB, 0, stream>>>(x2_32, W_p3, b_p3, out);
}
